// CustomMoEBranch_56547539419725
// MI455X (gfx1250) — matrix operand order repaired and checked by reading, not compiled, not run on hardware
//
#include <hip/hip_runtime.h>
#include <hip/hip_bf16.h>

// ---------- types ----------
typedef _Float16 half16 __attribute__((ext_vector_type(16)));
typedef _Float16 half8  __attribute__((ext_vector_type(8)));
typedef float    float8 __attribute__((ext_vector_type(8)));

#define ST 72    // hat (transposed convA act) row stride in halves: 144 B, 16B-multiple
#define SA 200   // wbs row stride in halves: 400 B, 16B-multiple

// ============================================================
// Kernel 1: STFT magnitude -> mean over frames  (pooled 64x129)
// one block per batch, 256 threads, table-driven DFT in LDS
// ============================================================
__global__ void stft_pool_kernel(const float* __restrict__ x,
                                 float* __restrict__ pooled) {
    __shared__ float xp[4352];
    __shared__ float win[256];
    __shared__ float ct[256];
    __shared__ float st[256];
    const int b = blockIdx.x;
    const int tid = threadIdx.x;

    for (int i = tid; i < 4352; i += 256) {
        int j = i - 128;
        if (j < 0) j = -j;
        else if (j > 4095) j = 8190 - j;
        xp[i] = x[(size_t)b * 4096 + j];
    }
    {
        float ang = 6.28318530717958647692f * (float)tid * (1.0f / 256.0f);
        float c = cosf(ang), s = sinf(ang);
        ct[tid] = c;
        st[tid] = s;
        win[tid] = 0.5f - 0.5f * c;
    }
    __syncthreads();

    if (tid < 129) {
        float acc = 0.0f;
        for (int f = 0; f < 65; ++f) {
            const float* fr = xp + f * 64;
            float re = 0.0f, im = 0.0f;
            for (int n = 0; n < 256; ++n) {
                float s = fr[n] * win[n];
                int k = (tid * n) & 255;
                re += s * ct[k];
                im += s * st[k];
            }
            acc += sqrtf(re * re + im * im);
        }
        pooled[b * 129 + tid] = acc * (1.0f / 65.0f);
    }
}

// ============================================================
// Kernel 2: gating MLP + top-2 + softmax  (one block per batch)
// ============================================================
__global__ void gate_kernel(const float* __restrict__ pooled,
                            const float* __restrict__ Wg1, const float* __restrict__ bg1,
                            const float* __restrict__ Wg2, const float* __restrict__ bg2,
                            const float* __restrict__ Wg3, const float* __restrict__ bg3,
                            int* __restrict__ einds, float* __restrict__ ewts) {
    __shared__ float pr[129];
    __shared__ float h1[256];
    __shared__ float h2[128];
    __shared__ float lg[8];
    const int b = blockIdx.x;
    const int tid = threadIdx.x;

    if (tid < 129) pr[tid] = pooled[b * 129 + tid];
    __syncthreads();

    for (int o = tid; o < 256; o += 128) {
        float a = bg1[o];
        for (int i = 0; i < 129; ++i) a += pr[i] * Wg1[i * 256 + o];
        h1[o] = fmaxf(a, 0.0f);
    }
    __syncthreads();

    if (tid < 128) {
        float a = bg2[tid];
        for (int i = 0; i < 256; ++i) a += h1[i] * Wg2[i * 128 + tid];
        h2[tid] = fmaxf(a, 0.0f);
    }
    __syncthreads();

    if (tid < 8) {
        float a = bg3[tid];
        for (int i = 0; i < 128; ++i) a += h2[i] * Wg3[i * 8 + tid];
        lg[tid] = a;
    }
    __syncthreads();

    if (tid == 0) {
        int i0 = 0; float v0 = lg[0];
        for (int i = 1; i < 8; ++i) if (lg[i] > v0) { v0 = lg[i]; i0 = i; }
        int i1 = -1; float v1 = -3.4e38f;
        for (int i = 0; i < 8; ++i) { if (i == i0) continue; if (lg[i] > v1) { v1 = lg[i]; i1 = i; } }
        float e1 = expf(v1 - v0);
        float inv = 1.0f / (1.0f + e1);
        einds[b * 2 + 0] = i0;  einds[b * 2 + 1] = i1;
        ewts[b * 2 + 0] = inv;  ewts[b * 2 + 1] = e1 * inv;
    }
}

// ============================================================
// WMMA fragment loads — contiguous 16B LDS reads, no packing.
// A (16x32, M=oc): lane L holds row M=L&15; element e holds
//   K = e + (e&8) + 8*(L>>4)  ->  two runs of 8 contiguous K.
// B (32x16, KxN): lane L holds col N=L&15; element e holds
//   K = e + (e&8) + 8*(L>>4), as for A. With hat[pos][ch] transposed staging the
//   16 elements are two runs of 16 contiguous bytes.
// K-chunk c of 6: tap = c>>1, ic0 = (c&1)*32.
// ============================================================
__device__ __forceinline__ half16 load_afrag(const _Float16* wbs, int oc_base,
                                             int chunk, int lane) {
    const int idx = lane & 15, hf = lane >> 4;
    const int tap = chunk >> 1, ic0 = (chunk & 1) * 32;
    const _Float16* p = wbs + (oc_base + idx) * SA + tap * 64 + ic0 + 8 * hf;
    half8 lo = *(const half8*)p;          // K = 8hf .. 8hf+7
    half8 hi = *(const half8*)(p + 16);   // K = 16+8hf .. 23+8hf
    return __builtin_shufflevector(lo, hi, 0, 1, 2, 3, 4, 5, 6, 7,
                                   8, 9, 10, 11, 12, 13, 14, 15);
}

__device__ __forceinline__ half16 load_bfrag(const _Float16* hat, int l0,
                                             int chunk, int lane) {
    const int idx = lane & 15, hf = lane >> 4;
    const int tap = chunk >> 1, ic0 = (chunk & 1) * 32;
    const int col = 2 * (l0 + idx) + tap;               // convA position
    const _Float16* p = hat + col * ST + ic0 + 8 * hf;  // ch 8hf..8hf+7, 16+8hf..23+8hf
    half8 lo = *(const half8*)p;
    half8 hi = *(const half8*)(p + 16);
    return __builtin_shufflevector(lo, hi, 0, 1, 2, 3, 4, 5, 6, 7,
                                   8, 9, 10, 11, 12, 13, 14, 15);
}

// ============================================================
// Kernel 3: fused convA(relu) -> convB(relu) implicit GEMM on
// v_wmma_f32_16x16x32_f16, gathered over the top-2 experts.
// grid = (16 pos-tiles, 3 branches, 64 batches), 128 threads.
// ============================================================
__global__ void __launch_bounds__(128)
moe_conv_kernel(const float* __restrict__ x,
                const float* __restrict__ wa3, const float* __restrict__ ba3,
                const float* __restrict__ wb3, const float* __restrict__ bb3,
                const float* __restrict__ wa5, const float* __restrict__ ba5,
                const float* __restrict__ wb5, const float* __restrict__ bb5,
                const float* __restrict__ wa7, const float* __restrict__ ba7,
                const float* __restrict__ wb7, const float* __restrict__ bb7,
                const int* __restrict__ einds, const float* __restrict__ ewts,
                float* __restrict__ out) {
    __shared__ __align__(16) _Float16 hat[129 * ST]; // convA act, TRANSPOSED [pos][ch]
    __shared__ __align__(16) _Float16 wbs[64 * SA];  // convB weights [oc][tap*64+ic]
    __shared__ float bbs[64];                        // convB bias

    const int tid    = threadIdx.x;
    const int lane   = tid & 31;
    const int wv     = tid >> 5;
    const int ptile  = blockIdx.x;
    const int branch = blockIdx.y;
    const int b      = blockIdx.z;
    const int p0     = ptile * 64;
    const int hq0    = 2 * p0 - 1;
    const int ksz    = 3 + 2 * branch;
    const int pad    = ksz >> 1;

    const float* wa = (branch == 0) ? wa3 : (branch == 1) ? wa5 : wa7;
    const float* ba = (branch == 0) ? ba3 : (branch == 1) ? ba5 : ba7;
    const float* wb = (branch == 0) ? wb3 : (branch == 1) ? wb5 : wb7;
    const float* bb = (branch == 0) ? bb3 : (branch == 1) ? bb5 : bb7;

    const float* xrow = x + (size_t)b * 4096;
    {   // pull the x tile toward the caches (global_prefetch_b8)
        int xi = 4 * p0 + tid * 2;
        if (xi > 4095) xi = 4095;
        __builtin_prefetch(xrow + xi, 0, 1);
    }

    const int oc_base = wv * 16;
    float8 acc_out[4];
#pragma unroll
    for (int nt = 0; nt < 4; ++nt)
#pragma unroll
        for (int r = 0; r < 8; ++r) acc_out[nt][r] = 0.0f;

    for (int j = 0; j < 2; ++j) {        // the two selected experts
        const int   e  = einds[b * 2 + j];
        const float wj = ewts[b * 2 + j];

        __syncthreads();                 // previous-LDS readers done

        // ---- stage convA (relu) transposed: hat[q][ch], paired b32 stores ----
        for (int t = tid; t < 129 * 32; t += 128) {
            int q  = t >> 5;             // 0..128  (global pos hq0+q)
            int ch = (t & 31) * 2;       // even channel
            int hq = hq0 + q;
            float v0 = 0.0f, v1 = 0.0f;
            if (hq >= 0 && hq < 2048) {
                v0 = ba[e * 64 + ch];
                v1 = ba[e * 64 + ch + 1];
                int xbase = 2 * hq - pad;
                for (int h = 0; h < ksz; ++h) {
                    int xi = xbase + h;
                    if (xi >= 0 && xi < 4096) {
                        float xv = xrow[xi];
                        v0 += wa[(e * 64 + ch) * ksz + h] * xv;
                        v1 += wa[(e * 64 + ch + 1) * ksz + h] * xv;
                    }
                }
                v0 = fmaxf(v0, 0.0f);
                v1 = fmaxf(v1, 0.0f);
            }
            union { _Float16 h[2]; unsigned u; } pk;
            pk.h[0] = (_Float16)v0;
            pk.h[1] = (_Float16)v1;
            *(unsigned*)&hat[q * ST + ch] = pk.u;   // ST,ch even -> 4B aligned
        }
        // ---- stage convB weights: wbs[oc][tap*64+ic], paired b32 stores ----
        for (int t = tid; t < 64 * 96; t += 128) {
            int oc = t / 96, r2 = (t - oc * 96) * 2;
            union { _Float16 h[2]; unsigned u; } pk;
#pragma unroll
            for (int s = 0; s < 2; ++s) {
                int rem = r2 + s, tap = rem >> 6, ic = rem & 63;
                pk.h[s] = (_Float16)wb[((size_t)(e * 64 + oc) * 64 + ic) * 3 + tap];
            }
            *(unsigned*)&wbs[oc * SA + r2] = pk.u;  // SA even -> 4B aligned
        }
        if (tid < 64) bbs[tid] = bb[e * 64 + tid];
        __syncthreads();

        // ---- implicit GEMM: M=16 oc, N=16 pos, K=192 (6 x 32) ----
        const int hf = lane >> 4;
#pragma unroll
        for (int nt = 0; nt < 4; ++nt) {
            float8 c;
#pragma unroll
            for (int r = 0; r < 8; ++r) c[r] = bbs[oc_base + r + 8 * hf];
#pragma unroll
            for (int ch = 0; ch < 6; ++ch) {
                half16 a  = load_afrag(wbs, oc_base, ch, lane);
                half16 bf = load_bfrag(hat, nt * 16, ch, lane);
                c = __builtin_amdgcn_wmma_f32_16x16x32_f16(
                        false, a, false, bf, (short)0, c, false, false);
            }
#pragma unroll
            for (int r = 0; r < 8; ++r)
                acc_out[nt][r] += wj * fmaxf(c[r], 0.0f);
        }
    }

    // ---- store: out[b][branch*64 + oc][p0 + l], each element once ----
    const int hf  = lane >> 4;
    const int idx = lane & 15;
#pragma unroll
    for (int nt = 0; nt < 4; ++nt) {
#pragma unroll
        for (int r = 0; r < 8; ++r) {
            int oc = oc_base + r + 8 * hf;
            int p  = p0 + nt * 16 + idx;
            out[((size_t)b * 192 + branch * 64 + oc) * 1024 + p] = acc_out[nt][r];
        }
    }
}

// ============================================================
// launch
// ============================================================
extern "C" void kernel_launch(void* const* d_in, const int* in_sizes, int n_in,
                              void* d_out, int out_size, void* d_ws, size_t ws_size,
                              hipStream_t stream) {
    const float* x   = (const float*)d_in[0];
    const float* Wg1 = (const float*)d_in[1];
    const float* bg1 = (const float*)d_in[2];
    const float* Wg2 = (const float*)d_in[3];
    const float* bg2 = (const float*)d_in[4];
    const float* Wg3 = (const float*)d_in[5];
    const float* bg3 = (const float*)d_in[6];
    const float* wa3 = (const float*)d_in[7];
    const float* ba3 = (const float*)d_in[8];
    const float* wb3 = (const float*)d_in[9];
    const float* bb3 = (const float*)d_in[10];
    const float* wa5 = (const float*)d_in[11];
    const float* ba5 = (const float*)d_in[12];
    const float* wb5 = (const float*)d_in[13];
    const float* bb5 = (const float*)d_in[14];
    const float* wa7 = (const float*)d_in[15];
    const float* ba7 = (const float*)d_in[16];
    const float* wb7 = (const float*)d_in[17];
    const float* bb7 = (const float*)d_in[18];

    float* pooled = (float*)d_ws;                          // 64*129 f32
    int*   einds  = (int*)((char*)d_ws + 33024);           // 64*2 int
    float* ewts   = (float*)((char*)d_ws + 33536);         // 64*2 f32

    stft_pool_kernel<<<64, 256, 0, stream>>>(x, pooled);
    gate_kernel<<<64, 128, 0, stream>>>(pooled, Wg1, bg1, Wg2, bg2, Wg3, bg3,
                                        einds, ewts);
    dim3 grid(16, 3, 64);
    moe_conv_kernel<<<grid, 128, 0, stream>>>(x,
        wa3, ba3, wb3, bb3, wa5, ba5, wb5, bb5, wa7, ba7, wb7, bb7,
        einds, ewts, (float*)d_out);
}
